// RNTensorN_32272384262685
// MI455X (gfx1250) — matrix operand order repaired and checked by reading, not compiled, not run on hardware
//
#include <hip/hip_runtime.h>
#include <hip/hip_bf16.h>
#include <cstdint>
#include <cstddef>

// Problem constants (from reference): L=2048, D=256, C=5
#define D_      256
#define TWOD    512
#define PAIRS   64          // pairs (output rows) per workgroup in level kernel
#define LEAVES  2048
#define DEPTH   11
#define NNODES  4095        // 2048+1024+...+1
#define NCLS    5

typedef __attribute__((ext_vector_type(16))) __bf16 v16bf;
typedef __attribute__((ext_vector_type(8)))  float  v8f;

union Frag16 { uint4 q[2]; unsigned u[8]; v16bf v; };

__device__ __forceinline__ unsigned pk_bf16(float a, float b) {
  // round-to-nearest-even f32 -> packed bf16x2 (a in low half, b in high half)
  unsigned ua = __float_as_uint(a), ub = __float_as_uint(b);
  ua = (ua + 0x7FFFu + ((ua >> 16) & 1u)) >> 16;
  ub = (ub + 0x7FFFu + ((ub >> 16) & 1u)) & 0xFFFF0000u;
  return ua | ub;
}

// ---------------------------------------------------------------------------
// f32 -> bf16 conversion (packed, 2 elements/thread)
// ---------------------------------------------------------------------------
__global__ __launch_bounds__(256) void cvt_bf16_kernel(
    const float* __restrict__ src, unsigned* __restrict__ dst, int npairs2) {
  int i = blockIdx.x * 256 + threadIdx.x;
  if (i < npairs2) {
    float2 f = ((const float2*)src)[i];
    dst[i] = pk_bf16(f.x, f.y);
  }
}

// ---------------------------------------------------------------------------
// h0 = emb[token_ids]   (2048 blocks x 256 threads)
// ---------------------------------------------------------------------------
__global__ __launch_bounds__(256) void embed_kernel(
    const int* __restrict__ tok, const float* __restrict__ emb,
    float* __restrict__ h0) {
  int i = blockIdx.x, j = threadIdx.x;
  h0[(size_t)i * D_ + j] = emb[(size_t)tok[i] * D_ + j];
}

// ---------------------------------------------------------------------------
// One tree level:
//   tens[n,a] = sum_{k,b} c[n,k]*c[n,b]*Vb[k*131072 + a*512 + b]
//   h_out[n,a] = tanh(tens + sum_b c[n,b]*Wb[a*512+b] + W_b[a])
// c = h_in viewed as (n_pairs, 512) (pairs are adjacent rows -> no gather).
// WG: 256 threads = 8 waves; tile: 64 pairs x 256 outputs.
// Wave w owns a-tiles [w*32, w*32+16) and [w*32+16, w*32+32), all 4 pair-tiles.
// ---------------------------------------------------------------------------
__global__ __launch_bounds__(256) void level_kernel(
    const unsigned short* __restrict__ Vb,   // 512*512*256 bf16 (same layout as f32 V)
    const unsigned short* __restrict__ Wb,   // 256*512 bf16 (W_w row-major)
    const float* __restrict__ W_b,           // 256
    const float* __restrict__ h_in,          // (2*n_pairs, 256) f32 == c (n_pairs, 512)
    float* __restrict__ h_out,               // (n_pairs, 256) f32
    int n_pairs) {
  __shared__ unsigned short c16[PAIRS * TWOD];   // 64 KB of 320 KB WGP LDS

  const int tid  = threadIdx.x;
  const int lane = tid & 31;
  const int w    = tid >> 5;
  const int m    = lane & 15;     // row (A/M) / col (B/N) within 16
  const int g    = lane >> 4;     // lane-group (K-half select)
  const int a0   = w * 32;        // this wave's output-feature base
  const int pairBase = blockIdx.x * PAIRS;

  // Stage c into LDS as bf16 (zero-fill rows past n_pairs).
  {
    unsigned* c32 = (unsigned*)c16;
    for (int i = tid; i < PAIRS * (TWOD / 2); i += 256) {
      int r = i / (TWOD / 2), col2 = i % (TWOD / 2);
      int pair = pairBase + r;
      unsigned pk = 0u;
      if (pair < n_pairs) {
        float2 f = ((const float2*)h_in)[(size_t)pair * (TWOD / 2) + col2];
        pk = pk_bf16(f.x, f.y);
      }
      c32[i] = pk;
    }
  }
  __syncthreads();

  const unsigned* cpk = (const unsigned*)c16;

  v8f zero = {0.f, 0.f, 0.f, 0.f, 0.f, 0.f, 0.f, 0.f};
  v8f acc[4][2];
#pragma unroll
  for (int t = 0; t < 4; ++t) { acc[t][0] = zero; acc[t][1] = zero; }

  // ---- main K loop: k over 512, b in chunks of 32 (WMMA K) ----
#pragma unroll 1
  for (int k = 0; k < TWOD; ++k) {
    unsigned spk[4];
#pragma unroll
    for (int t = 0; t < 4; ++t) {
      unsigned s = (unsigned)c16[(t * 16 + m) * TWOD + k];  // bf16 bits of c[row,k]
      spk[t] = (s << 16) | s;                               // replicate for pk-mul
    }
    const unsigned short* vk = Vb + (size_t)k * (TWOD * D_);

#pragma unroll 1
    for (int bc = 0; bc < 16; ++bc) {
      const int b0 = bc * 32;
      // B fragments (V tile), reused by all 4 pair-tiles: 16 contiguous bf16/lane
      Frag16 Bf[2];
#pragma unroll
      for (int at = 0; at < 2; ++at) {
        const unsigned short* p = vk + (size_t)(a0 + at * 16 + m) * TWOD + b0 + g * 16;
        Bf[at].q[0] = ((const uint4*)p)[0];
        Bf[at].q[1] = ((const uint4*)p)[1];
      }
#pragma unroll
      for (int t = 0; t < 4; ++t) {
        // A fragment = c[row, k] * c[row, b0..b0+31] in bf16 (packed pk-mul)
        const unsigned* crow = cpk + (t * 16 + m) * (TWOD / 2);
        uint4 lo = ((const uint4*)(crow + (b0 + g * 16) / 2))[0];
        uint4 hi = ((const uint4*)(crow + (b0 + g * 16 + 8) / 2))[0];
        Frag16 Af;
        asm volatile("v_pk_mul_bf16 %0, %1, %2" : "=v"(Af.u[0]) : "v"(lo.x), "v"(spk[t]));
        asm volatile("v_pk_mul_bf16 %0, %1, %2" : "=v"(Af.u[1]) : "v"(lo.y), "v"(spk[t]));
        asm volatile("v_pk_mul_bf16 %0, %1, %2" : "=v"(Af.u[2]) : "v"(lo.z), "v"(spk[t]));
        asm volatile("v_pk_mul_bf16 %0, %1, %2" : "=v"(Af.u[3]) : "v"(lo.w), "v"(spk[t]));
        asm volatile("v_pk_mul_bf16 %0, %1, %2" : "=v"(Af.u[4]) : "v"(hi.x), "v"(spk[t]));
        asm volatile("v_pk_mul_bf16 %0, %1, %2" : "=v"(Af.u[5]) : "v"(hi.y), "v"(spk[t]));
        asm volatile("v_pk_mul_bf16 %0, %1, %2" : "=v"(Af.u[6]) : "v"(hi.z), "v"(spk[t]));
        asm volatile("v_pk_mul_bf16 %0, %1, %2" : "=v"(Af.u[7]) : "v"(hi.w), "v"(spk[t]));
        acc[t][0] = __builtin_amdgcn_wmma_f32_16x16x32_bf16(
            false, Af.v, false, Bf[0].v, (short)0, acc[t][0], false, false);
        acc[t][1] = __builtin_amdgcn_wmma_f32_16x16x32_bf16(
            false, Af.v, false, Bf[1].v, (short)0, acc[t][1], false, false);
      }
    }
  }

  // ---- folded W pass: + c @ W_w^T  (A = c, B = Wb) ----
#pragma unroll 1
  for (int bc = 0; bc < 16; ++bc) {
    const int b0 = bc * 32;
    Frag16 Bf[2];
#pragma unroll
    for (int at = 0; at < 2; ++at) {
      const unsigned short* p = Wb + (size_t)(a0 + at * 16 + m) * TWOD + b0 + g * 16;
      Bf[at].q[0] = ((const uint4*)p)[0];
      Bf[at].q[1] = ((const uint4*)p)[1];
    }
#pragma unroll
    for (int t = 0; t < 4; ++t) {
      const unsigned* crow = cpk + (t * 16 + m) * (TWOD / 2);
      Frag16 Af;
      Af.q[0] = ((const uint4*)(crow + (b0 + g * 16) / 2))[0];
      Af.q[1] = ((const uint4*)(crow + (b0 + g * 16 + 8) / 2))[0];
      acc[t][0] = __builtin_amdgcn_wmma_f32_16x16x32_bf16(
          false, Af.v, false, Bf[0].v, (short)0, acc[t][0], false, false);
      acc[t][1] = __builtin_amdgcn_wmma_f32_16x16x32_bf16(
          false, Af.v, false, Bf[1].v, (short)0, acc[t][1], false, false);
    }
  }

  // ---- epilogue: bias + tanh, store (C layout: VGPR v -> M=v (lanes<16) / v+8) ----
#pragma unroll
  for (int t = 0; t < 4; ++t) {
#pragma unroll
    for (int at = 0; at < 2; ++at) {
      const int a = a0 + at * 16 + m;
      const float bias = W_b[a];
#pragma unroll
      for (int v = 0; v < 8; ++v) {
        int row = t * 16 + v + 8 * g;
        int pair = pairBase + row;
        if (pair < n_pairs)
          h_out[(size_t)pair * D_ + a] = tanhf(acc[t][at][v] + bias);
      }
    }
  }
}

// ---------------------------------------------------------------------------
// Classifier: one wave per node. Computes post-order rank analytically:
// rank(l,p) = sum_{j=l+1..depth} [(p>>(j-1-l))&1] * (2^j - 1) + 2^(l+1) - 2
// then logits = rep @ Ws_w^T + Ws_b, softmax over 5, scatter to out[rank].
// ---------------------------------------------------------------------------
__global__ __launch_bounds__(256) void classify_kernel(
    const float* __restrict__ reps,   // (4095, 256) level-major
    const float* __restrict__ Ws_w,   // (5, 256)
    const float* __restrict__ Ws_b,   // (5,)
    float* __restrict__ out) {        // (4095, 5)
  int wv = (blockIdx.x * blockDim.x + threadIdx.x) >> 5;
  int lane = threadIdx.x & 31;
  if (wv >= NNODES) return;

  // decompose level-major node index -> (level l, position p)
  int idx = wv, l = 0, sz = LEAVES;
  while (idx >= sz) { idx -= sz; ++l; sz >>= 1; }
  int p = idx;

  int base = 0;
  for (int j = l + 1; j <= DEPTH; ++j)
    if ((p >> (j - 1 - l)) & 1) base += (1 << j) - 1;
  int rank = base + (1 << (l + 1)) - 2;

  const float* r = reps + (size_t)wv * D_;
  float4 f0 = ((const float4*)r)[lane * 2 + 0];
  float4 f1 = ((const float4*)r)[lane * 2 + 1];

  float logit[NCLS];
#pragma unroll
  for (int c = 0; c < NCLS; ++c) {
    const float* wr = Ws_w + c * D_;
    float4 w0 = ((const float4*)wr)[lane * 2 + 0];
    float4 w1 = ((const float4*)wr)[lane * 2 + 1];
    float s = f0.x * w0.x + f0.y * w0.y + f0.z * w0.z + f0.w * w0.w +
              f1.x * w1.x + f1.y * w1.y + f1.z * w1.z + f1.w * w1.w;
#pragma unroll
    for (int o = 16; o > 0; o >>= 1) s += __shfl_xor(s, o, 32);
    logit[c] = s + Ws_b[c];
  }
  if (lane == 0) {
    float mx = logit[0];
#pragma unroll
    for (int c = 1; c < NCLS; ++c) mx = fmaxf(mx, logit[c]);
    float e[NCLS], sum = 0.f;
#pragma unroll
    for (int c = 0; c < NCLS; ++c) { e[c] = expf(logit[c] - mx); sum += e[c]; }
    float inv = 1.f / sum;
#pragma unroll
    for (int c = 0; c < NCLS; ++c) out[(size_t)rank * NCLS + c] = e[c] * inv;
  }
}

// ---------------------------------------------------------------------------
// Host launch. Workspace layout (needs ~132.3 MiB):
//   [0, 128MiB)            Vb  : V in bf16 (67,108,864 elems) -> fits 192MB L2
//   [+256KiB)              Wb  : W_w in bf16 (131,072 elems)
//   [+4.0MiB)              reps: all node reps f32 (4095 x 256), level-major
// ---------------------------------------------------------------------------
extern "C" void kernel_launch(void* const* d_in, const int* in_sizes, int n_in,
                              void* d_out, int out_size, void* d_ws, size_t ws_size,
                              hipStream_t stream) {
  (void)in_sizes; (void)n_in; (void)out_size; (void)ws_size;
  const int*   tok  = (const int*)d_in[0];
  const float* emb  = (const float*)d_in[1];
  const float* V    = (const float*)d_in[2];
  const float* W_w  = (const float*)d_in[3];
  const float* W_b  = (const float*)d_in[4];
  const float* Ws_w = (const float*)d_in[5];
  const float* Ws_b = (const float*)d_in[6];

  const size_t NV = (size_t)TWOD * TWOD * D_;    // 67,108,864
  const size_t NW = (size_t)D_ * TWOD;           // 131,072
  unsigned short* Vb = (unsigned short*)d_ws;
  unsigned short* Wb = Vb + NV;
  float* reps = (float*)(Wb + NW);

  // 1) bf16 conversions (V becomes L2-resident for the 11 level passes)
  {
    int npairsV = (int)(NV / 2);
    cvt_bf16_kernel<<<(npairsV + 255) / 256, 256, 0, stream>>>(V, (unsigned*)Vb, npairsV);
    int npairsW = (int)(NW / 2);
    cvt_bf16_kernel<<<(npairsW + 255) / 256, 256, 0, stream>>>(W_w, (unsigned*)Wb, npairsW);
  }

  // 2) leaves
  embed_kernel<<<LEAVES, 256, 0, stream>>>(tok, emb, reps);

  // 3) 11 sequential tree levels (stream order provides the dependency)
  int nrows = LEAVES;
  size_t inOff = 0;
  while (nrows > 1) {
    int np = nrows >> 1;
    size_t outOff = inOff + (size_t)nrows * D_;
    int grid = (np + PAIRS - 1) / PAIRS;
    level_kernel<<<grid, 256, 0, stream>>>(Vb, Wb, W_b,
                                           reps + inOff, reps + outOff, np);
    inOff = outOff;
    nrows = np;
  }

  // 4) post-order permute + classify + softmax
  classify_kernel<<<(NNODES + 7) / 8, 256, 0, stream>>>(reps, Ws_w, Ws_b, (float*)d_out);
}
